// SparseMoE_74019466379779
// MI455X (gfx1250) — matrix operand order repaired and checked by reading, not compiled, not run on hardware
//
#include <hip/hip_runtime.h>
#include <stdint.h>

// ---------------- problem constants (from reference) ----------------
#define D_DIM 1024
#define H_DIM 4096
#define E_NUM 8
#define T_TOK 4096   // B*S = 2*2048
#define HC    512    // H-chunk staged in LDS per iteration
#define MT    16     // token-tile rows (M of WMMA)

// ---------------- types for CDNA5 WMMA ----------------
typedef __attribute__((ext_vector_type(16))) __bf16        bf16x16;
typedef __attribute__((ext_vector_type(8)))  float         f32x8;
typedef __attribute__((ext_vector_type(4)))  unsigned int  u32x4;

union Frag {
  bf16x16 v;
  u32x4   q[2];
};

__device__ __forceinline__ unsigned short f2bf(float f) {
  // round-to-nearest-even fp32 -> bf16 bit pattern
  unsigned u = __float_as_uint(f);
  unsigned r = u + 0x7FFFu + ((u >> 16) & 1u);
  return (unsigned short)(r >> 16);
}

#define WMMA_BF16(A, B, C)                                                     \
  __builtin_amdgcn_wmma_f32_16x16x32_bf16(false, (A).v, false, (B).v,          \
                                          (short)0, (C), false, false)

// ---------------- 1) weight transpose + fp32->bf16 ----------------
// src: [E][R][C] fp32, dst: [E][C][R] bf16 (K-contiguous for B fragments)
__global__ void moe_w_transcvt(const float* __restrict__ src,
                               unsigned short* __restrict__ dst,
                               int R, int C) {
  __shared__ float tile[32][33];
  const int e = blockIdx.z;
  const long base = (long)e * R * C;
  const int c0 = blockIdx.x * 32, r0 = blockIdx.y * 32;
  for (int j = threadIdx.y; j < 32; j += 8)
    tile[j][threadIdx.x] = src[base + (long)(r0 + j) * C + (c0 + threadIdx.x)];
  __syncthreads();
  for (int j = threadIdx.y; j < 32; j += 8)
    dst[base + (long)(c0 + j) * R + (r0 + threadIdx.x)] = f2bf(tile[threadIdx.x][j]);
}

// ---------------- 2) router: logits -> top2 -> softmax gates ----------------
__global__ void moe_router(const float* __restrict__ x,
                           const float* __restrict__ rw,
                           const float* __restrict__ rb,
                           int* __restrict__ topk_idx,
                           float* __restrict__ topk_gate) {
  const int lane = threadIdx.x & 31;
  const int tok  = blockIdx.x * 8 + (threadIdx.x >> 5);   // one wave32 per token
  float acc[E_NUM];
#pragma unroll
  for (int e = 0; e < E_NUM; ++e) acc[e] = 0.f;
  const float* xr = x + (long)tok * D_DIM;
  for (int d = lane; d < D_DIM; d += 32) {
    const float xv = xr[d];
    const float* w = rw + d * E_NUM;   // router_w is [D][E], E contiguous
#pragma unroll
    for (int e = 0; e < E_NUM; ++e) acc[e] += xv * w[e];
  }
#pragma unroll
  for (int e = 0; e < E_NUM; ++e) {
#pragma unroll
    for (int off = 16; off > 0; off >>= 1)
      acc[e] += __shfl_xor(acc[e], off, 32);
  }
  if (lane == 0) {
#pragma unroll
    for (int e = 0; e < E_NUM; ++e) acc[e] += rb[e];
    int i0 = 0;
    for (int e = 1; e < E_NUM; ++e) if (acc[e] > acc[i0]) i0 = e;   // strict >: lowest idx on tie
    int i1 = (i0 == 0) ? 1 : 0;
    for (int e = 0; e < E_NUM; ++e) if (e != i0 && acc[e] > acc[i1]) i1 = e;
    const float l0 = acc[i0], l1 = acc[i1];
    const float e1  = expf(l1 - l0);          // l0 >= l1; softmax over {l0,l1}
    const float inv = 1.f / (1.f + e1);
    topk_idx[2 * tok]      = i0;
    topk_idx[2 * tok + 1]  = i1;
    topk_gate[2 * tok]     = inv;
    topk_gate[2 * tok + 1] = e1 * inv;
  }
}

// ---------------- 3) count / scan / scatter into per-expert lists ----------------
__global__ void moe_zero32(int* __restrict__ p, int n) {
  int i = blockIdx.x * 64 + threadIdx.x;
  if (i < n) p[i] = 0;
}

__global__ void moe_count(const int* __restrict__ topk_idx, int* __restrict__ cnt) {
  int t = blockIdx.x * 256 + threadIdx.x;
  if (t < T_TOK) {
    atomicAdd(&cnt[topk_idx[2 * t]], 1);
    atomicAdd(&cnt[topk_idx[2 * t + 1]], 1);
  }
}

__global__ void moe_scan(const int* __restrict__ cnt, int* __restrict__ offs) {
  if (threadIdx.x == 0 && blockIdx.x == 0) {
    int s = 0;
    for (int e = 0; e < E_NUM; ++e) { offs[e] = s; s += cnt[e]; }
  }
}

__global__ void moe_scatter(const int* __restrict__ topk_idx,
                            const float* __restrict__ topk_gate,
                            const int* __restrict__ offs,
                            int* __restrict__ cnt2,
                            int* __restrict__ rows_token,
                            float* __restrict__ rows_gate,
                            int* __restrict__ slot_of) {
  int t = blockIdx.x * 256 + threadIdx.x;
  if (t >= T_TOK) return;
#pragma unroll
  for (int j = 0; j < 2; ++j) {
    const int e = topk_idx[2 * t + j];
    const int p = atomicAdd(&cnt2[e], 1);
    const int row = offs[e] + p;
    rows_token[row] = t;
    rows_gate[row] = topk_gate[2 * t + j];
    slot_of[2 * t + j] = row;   // deterministic combine mapping
  }
}

// ---------------- 4) fused expert MLP: y = (relu(X@W1+b1)@W2 + b2) * gate ----------------
// grid: (T_TOK/MT tiles, E experts); 256 threads = 8 waves; wave32 WMMA bf16.
// Software-pipelined: next k-step fragments load while current step's two
// independent WMMAs (dual N-tiles sharing the A fragment) execute.
__launch_bounds__(256, 1)
__global__ void moe_expert_mlp(const float* __restrict__ x,
                               const unsigned short* __restrict__ wt1,  // [E][H][D] bf16
                               const unsigned short* __restrict__ wt2,  // [E][D][H] bf16
                               const float* __restrict__ b1,
                               const float* __restrict__ b2,
                               const int* __restrict__ cnt,
                               const int* __restrict__ offs,
                               const int* __restrict__ rows_token,
                               const float* __restrict__ rows_gate,
                               float* __restrict__ y_tmp) {
  __shared__ __align__(16) unsigned short xs[MT * D_DIM]; // 32 KB: X tile, bf16
  __shared__ __align__(16) unsigned short hs[MT * HC];    // 16 KB: h chunk, bf16
  __shared__ int tok[MT];

  const int e = blockIdx.y;
  const int cnt_e = cnt[e];
  const int tile0 = blockIdx.x * MT;
  if (tile0 >= cnt_e) return;                 // uniform per block
  const int off_e = offs[e];

  const int tid  = threadIdx.x;
  const int lane = tid & 31;
  const int wave = tid >> 5;

  if (tid < MT) {
    const int gr = tile0 + tid;
    tok[tid] = (gr < cnt_e) ? rows_token[off_e + gr] : -1;
  }
  __syncthreads();
  // gather routed token rows, fp32 -> bf16
  for (int i = tid; i < MT * D_DIM; i += 256) {
    const int r = i >> 10, c = i & (D_DIM - 1);
    const int t = tok[r];
    xs[i] = (t >= 0) ? f2bf(x[(long)t * D_DIM + c]) : (unsigned short)0;
  }

  const unsigned short* wt1e = wt1 + (size_t)e * H_DIM * D_DIM;
  const unsigned short* wt2e = wt2 + (size_t)e * D_DIM * H_DIM;
  const float* b1e = b1 + e * H_DIM;
  const float* b2e = b2 + e * D_DIM;

  // wave32 fragment geometry (m/n per ISA 7.12.2; K order is this code's contiguous k = 16h + s):
  const int rsel  = lane >> 4;       // 0: lanes 0-15, 1: lanes 16-31
  const int acol  = lane & 15;       // A: row index; B/C: column index
  const int akoff = rsel << 4;       // A frag: +16 K for upper lane half
  const int bkoff = rsel << 4;       // B frag: +16 K for upper lane half
  const int rb    = rsel * 8;        // C frag: rows rb..rb+7

  const f32x8 vzero = {0.f, 0.f, 0.f, 0.f, 0.f, 0.f, 0.f, 0.f};
  f32x8 yacc[8];
#pragma unroll
  for (int t = 0; t < 8; ++t) yacc[t] = vzero;

  for (int c = 0; c < H_DIM / HC; ++c) {
    __syncthreads();   // hs free (prev phase-2 done) + xs ready on first pass
    // ---- phase 1: hs = relu(X @ W1 + b1) for H columns [c*HC, c*HC+HC) ----
#pragma unroll
    for (int pair = 0; pair < 2; ++pair) {
      const int ncol0 = wave * 64 + pair * 32;            // two adjacent 16-col tiles
      const int hg0 = c * HC + ncol0 + acol;              // lane's H column, tile 0
      const int hg1 = hg0 + 16;                           // lane's H column, tile 1
      const unsigned short* arow = xs + acol * D_DIM + akoff;
      const unsigned short* br0  = wt1e + (size_t)hg0 * D_DIM + bkoff;
      const unsigned short* br1  = wt1e + (size_t)hg1 * D_DIM + bkoff;
      f32x8 h0 = vzero, h1 = vzero;
      Frag a, b0, b1f;
      a.q[0]   = *(const u32x4*)(arow);
      a.q[1]   = *(const u32x4*)(arow + 8);
      b0.q[0]  = *(const u32x4*)(br0);
      b0.q[1]  = *(const u32x4*)(br0 + 8);
      b1f.q[0] = *(const u32x4*)(br1);
      b1f.q[1] = *(const u32x4*)(br1 + 8);
      for (int kb = 32; kb < D_DIM; kb += 32) {
        Frag an, b0n, b1n;                                // prefetch next step
        an.q[0]  = *(const u32x4*)(arow + kb);
        an.q[1]  = *(const u32x4*)(arow + kb + 8);
        b0n.q[0] = *(const u32x4*)(br0 + kb);
        b0n.q[1] = *(const u32x4*)(br0 + kb + 8);
        b1n.q[0] = *(const u32x4*)(br1 + kb);
        b1n.q[1] = *(const u32x4*)(br1 + kb + 8);
        __builtin_prefetch(br0 + kb + 32, 0, 1);
        __builtin_prefetch(br1 + kb + 32, 0, 1);
        h0 = WMMA_BF16(a, b0, h0);                        // overlap with loads above
        h1 = WMMA_BF16(a, b1f, h1);
        a = an; b0 = b0n; b1f = b1n;
      }
      h0 = WMMA_BF16(a, b0, h0);
      h1 = WMMA_BF16(a, b1f, h1);
      // relu + bias, store bf16 into hs
      const float bias0 = b1e[hg0];
      const float bias1 = b1e[hg1];
      const int nc0 = ncol0 + acol;
#pragma unroll
      for (int r = 0; r < 8; ++r) {
        float v0 = h0[r] + bias0;
        float v1 = h1[r] + bias1;
        hs[(rb + r) * HC + nc0]      = f2bf(v0 > 0.f ? v0 : 0.f);
        hs[(rb + r) * HC + nc0 + 16] = f2bf(v1 > 0.f ? v1 : 0.f);
      }
    }
    __syncthreads();
    // ---- phase 2: yacc += h_chunk @ W2[c*HC : c*HC+HC, :] ----
#pragma unroll
    for (int p = 0; p < 4; ++p) {
      const int n0 = wave * 128 + p * 32 + acol;          // output D columns (two tiles)
      const unsigned short* arow = hs + acol * HC + akoff;
      const unsigned short* br0  = wt2e + (size_t)n0 * H_DIM + c * HC + bkoff;
      const unsigned short* br1  = wt2e + (size_t)(n0 + 16) * H_DIM + c * HC + bkoff;
      f32x8 acc0 = yacc[2 * p], acc1 = yacc[2 * p + 1];
      Frag a, b0, b1f;
      a.q[0]   = *(const u32x4*)(arow);
      a.q[1]   = *(const u32x4*)(arow + 8);
      b0.q[0]  = *(const u32x4*)(br0);
      b0.q[1]  = *(const u32x4*)(br0 + 8);
      b1f.q[0] = *(const u32x4*)(br1);
      b1f.q[1] = *(const u32x4*)(br1 + 8);
      for (int kb = 32; kb < HC; kb += 32) {
        Frag an, b0n, b1n;
        an.q[0]  = *(const u32x4*)(arow + kb);
        an.q[1]  = *(const u32x4*)(arow + kb + 8);
        b0n.q[0] = *(const u32x4*)(br0 + kb);
        b0n.q[1] = *(const u32x4*)(br0 + kb + 8);
        b1n.q[0] = *(const u32x4*)(br1 + kb);
        b1n.q[1] = *(const u32x4*)(br1 + kb + 8);
        __builtin_prefetch(br0 + kb + 32, 0, 1);
        __builtin_prefetch(br1 + kb + 32, 0, 1);
        acc0 = WMMA_BF16(a, b0, acc0);
        acc1 = WMMA_BF16(a, b1f, acc1);
        a = an; b0 = b0n; b1f = b1n;
      }
      acc0 = WMMA_BF16(a, b0, acc0);
      acc1 = WMMA_BF16(a, b1f, acc1);
      yacc[2 * p] = acc0;
      yacc[2 * p + 1] = acc1;
    }
  }
  // ---- epilogue: gate * (y + b2) -> per-(token,expert) row (no atomics) ----
#pragma unroll
  for (int t = 0; t < 8; ++t) {
    const int n = wave * 128 + t * 16 + acol;
    const float b2v = b2e[n];
#pragma unroll
    for (int r = 0; r < 8; ++r) {
      const int gr = tile0 + rb + r;
      if (gr < cnt_e) {
        const int rowg = off_e + gr;
        y_tmp[(long)rowg * D_DIM + n] = (yacc[t][r] + b2v) * rows_gate[rowg];
      }
    }
  }
}

// ---------------- 5) combine: out[t] = y_row(slot0) + y_row(slot1) ----------------
__global__ void moe_combine(const float* __restrict__ y_tmp,
                            const int* __restrict__ slot_of,
                            float* __restrict__ out) {
  const long i = (long)blockIdx.x * 256 + threadIdx.x;
  const int t = (int)(i >> 10);
  const int d = (int)(i & (D_DIM - 1));
  const float a = y_tmp[(long)slot_of[2 * t] * D_DIM + d];
  const float b = y_tmp[(long)slot_of[2 * t + 1] * D_DIM + d];
  out[i] = a + b;
}

// ---------------- launch ----------------
extern "C" void kernel_launch(void* const* d_in, const int* in_sizes, int n_in,
                              void* d_out, int out_size, void* d_ws, size_t ws_size,
                              hipStream_t stream) {
  const float* x  = (const float*)d_in[0];
  const float* rw = (const float*)d_in[1];
  const float* rb = (const float*)d_in[2];
  const float* w1 = (const float*)d_in[3];
  const float* b1 = (const float*)d_in[4];
  const float* w2 = (const float*)d_in[5];
  const float* b2 = (const float*)d_in[6];
  float* out = (float*)d_out;

  // scratch carve-out (~168 MB total; assumed <= ws_size)
  char* ws = (char*)d_ws;
  size_t o = 0;
  auto carve = [&](size_t bytes) -> char* {
    char* p = ws + o;
    o = (o + bytes + 255) & ~(size_t)255;
    return p;
  };
  unsigned short* wt1 = (unsigned short*)carve((size_t)E_NUM * H_DIM * D_DIM * 2);
  unsigned short* wt2 = (unsigned short*)carve((size_t)E_NUM * H_DIM * D_DIM * 2);
  int*   cnt        = (int*)carve(E_NUM * 4);
  int*   cnt2       = (int*)carve(E_NUM * 4);
  int*   offs       = (int*)carve(E_NUM * 4);
  int*   topk_idx   = (int*)carve((size_t)2 * T_TOK * 4);
  float* topk_gate  = (float*)carve((size_t)2 * T_TOK * 4);
  int*   rows_token = (int*)carve((size_t)2 * T_TOK * 4);
  float* rows_gate  = (float*)carve((size_t)2 * T_TOK * 4);
  int*   slot_of    = (int*)carve((size_t)2 * T_TOK * 4);
  float* y_tmp      = (float*)carve((size_t)2 * T_TOK * D_DIM * 4);
  (void)ws_size; (void)in_sizes; (void)n_in; (void)out_size;

  // 1) one-time weight transform (L2-resident bf16, K-contiguous)
  moe_w_transcvt<<<dim3(H_DIM / 32, D_DIM / 32, E_NUM), dim3(32, 8), 0, stream>>>(
      w1, wt1, D_DIM, H_DIM);
  moe_w_transcvt<<<dim3(D_DIM / 32, H_DIM / 32, E_NUM), dim3(32, 8), 0, stream>>>(
      w2, wt2, H_DIM, D_DIM);
  // 2) router (one wave per token)
  moe_router<<<T_TOK / 8, 256, 0, stream>>>(x, rw, rb, topk_idx, topk_gate);
  // 3) build per-expert token lists
  moe_zero32<<<1, 64, 0, stream>>>(cnt, E_NUM);
  moe_zero32<<<1, 64, 0, stream>>>(cnt2, E_NUM);
  moe_count<<<T_TOK / 256, 256, 0, stream>>>(topk_idx, cnt);
  moe_scan<<<1, 1, 0, stream>>>(cnt, offs);
  moe_scatter<<<T_TOK / 256, 256, 0, stream>>>(topk_idx, topk_gate, offs, cnt2,
                                               rows_token, rows_gate, slot_of);
  // 4) fused expert MLP (WMMA bf16); expert-major grid.y keeps weights hot in L2
  moe_expert_mlp<<<dim3(T_TOK / MT, E_NUM), 256, 0, stream>>>(
      x, wt1, wt2, b1, b2, cnt, offs, rows_token, rows_gate, y_tmp);
  // 5) deterministic combine of each token's two expert rows
  moe_combine<<<(T_TOK * D_DIM) / 256, 256, 0, stream>>>(y_tmp, slot_of, out);
}
